// TransformerDecoderLayer_43748536877313
// MI455X (gfx1250) — compile-verified, matrix operand order checked by reading, not run on hardware
//
#include <hip/hip_runtime.h>

// ---------------------------------------------------------------------------
// Transformer decoder layer (DETR-style) for MI455X / gfx1250.
// B=16, Q=100, S=8192, D=256, H=8, hd=32, DFF=1024.
// All GEMMs via v_wmma_f32_16x16x32_bf16 (fp32 accum). K/V cached in bf16
// (fits 192MB L2), V stored transposed [b,h,d,S] for contiguous B-fragments.
// Flash-attention online softmax kept in registers (C-layout row ownership).
// Attention softmax scale is folded into the Q-projection epilogue.
// ---------------------------------------------------------------------------

typedef __attribute__((ext_vector_type(16))) __bf16 v16bf;
typedef __attribute__((ext_vector_type(8)))  __bf16 v8bf;
typedef __attribute__((ext_vector_type(8)))  float  v8f;
typedef __attribute__((ext_vector_type(4)))  float  v4f;

#define D_MODEL 256
#define NHEADS  8
#define HEAD_DIM 32
#define ATTN_SCALE 0.17677669529663687f  // 32^-0.5

// ---- fragment builders (ISA 7.12.2 layouts, wave32) -----------------------
// A 16x32 bf16: lane(0..15)=row M, elems {c0..c0+7, c1..c1+7} are K indices,
// c0 = (lane<16 ? 0 : 8), c1 = (lane<16 ? 16 : 24). B (32x16) mirrored: lane
// holds a column, same K-chunk mapping.  C/D: VGPR j -> row j + (lane<16?0:8),
// col = lane&15.

__device__ __forceinline__ v16bf frag_from_bf16(const __bf16* rp, int c0, int c1) {
  v8bf a = *(const v8bf*)(rp + c0);
  v8bf b = *(const v8bf*)(rp + c1);
  v16bf r;
#pragma unroll
  for (int i = 0; i < 8; ++i) { r[i] = a[i]; r[8 + i] = b[i]; }
  return r;
}

__device__ __forceinline__ v16bf frag_from_f32(const float* rp, int c0, int c1) {
  v4f a0 = *(const v4f*)(rp + c0);
  v4f a1 = *(const v4f*)(rp + c0 + 4);
  v4f a2 = *(const v4f*)(rp + c1);
  v4f a3 = *(const v4f*)(rp + c1 + 4);
  v16bf r;
#pragma unroll
  for (int i = 0; i < 4; ++i) {
    r[i]      = (__bf16)a0[i];
    r[4 + i]  = (__bf16)a1[i];
    r[8 + i]  = (__bf16)a2[i];
    r[12 + i] = (__bf16)a3[i];
  }
  return r;
}

__device__ __forceinline__ v16bf frag_from_f32_add(const float* rp, const float* rp2,
                                                   int c0, int c1) {
  v4f a0 = *(const v4f*)(rp + c0)     + *(const v4f*)(rp2 + c0);
  v4f a1 = *(const v4f*)(rp + c0 + 4) + *(const v4f*)(rp2 + c0 + 4);
  v4f a2 = *(const v4f*)(rp + c1)     + *(const v4f*)(rp2 + c1);
  v4f a3 = *(const v4f*)(rp + c1 + 4) + *(const v4f*)(rp2 + c1 + 4);
  v16bf r;
#pragma unroll
  for (int i = 0; i < 4; ++i) {
    r[i]      = (__bf16)a0[i];
    r[4 + i]  = (__bf16)a1[i];
    r[8 + i]  = (__bf16)a2[i];
    r[12 + i] = (__bf16)a3[i];
  }
  return r;
}

#define WMMA_BF16(Afrag, Bfrag, Cfrag) \
  __builtin_amdgcn_wmma_f32_16x16x32_bf16(false, (Afrag), false, (Bfrag), \
                                          (short)0, (Cfrag), false, false)

// ---- weight pre-transpose: W[K][N] f32 -> Wt[N][K] bf16 -------------------
__global__ void wtrans_kernel(const float* __restrict__ W, __bf16* __restrict__ Wt,
                              int K, int N) {
  int i = blockIdx.x * blockDim.x + threadIdx.x;
  if (i >= K * N) return;
  int k = i / N, n = i - k * N;
  Wt[(size_t)n * K + k] = (__bf16)W[i];
}

// ---- layernorm over last dim (256), one row per block ---------------------
__global__ void ln_kernel(const float* __restrict__ x, const float* __restrict__ g,
                          const float* __restrict__ bta, float* __restrict__ y) {
  int row = blockIdx.x;
  int t = threadIdx.x;                     // 256 threads = 8 waves
  size_t base = (size_t)row * D_MODEL;
  float v = x[base + t];
  __shared__ float red[8];
  float s = v;
#pragma unroll
  for (int m = 16; m >= 1; m >>= 1) s += __shfl_xor(s, m, 32);
  if ((t & 31) == 0) red[t >> 5] = s;
  __syncthreads();
  float tot = 0.f;
#pragma unroll
  for (int i = 0; i < 8; ++i) tot += red[i];
  float mu = tot * (1.0f / D_MODEL);
  __syncthreads();
  float d = v - mu;
  s = d * d;
#pragma unroll
  for (int m = 16; m >= 1; m >>= 1) s += __shfl_xor(s, m, 32);
  if ((t & 31) == 0) red[t >> 5] = s;
  __syncthreads();
  tot = 0.f;
#pragma unroll
  for (int i = 0; i < 8; ++i) tot += red[i];
  float var = tot * (1.0f / D_MODEL);
  y[base + t] = d * rsqrtf(var + 1e-5f) * g[t] + bta[t];
}

// ---- generic WMMA GEMM: out = act(((A [+A2]) @ Wt^T + bias)*oscale) [+res]
// Block = 128 threads (4 waves); each wave computes a 16(M) x 32(N) tile.
// grid.x = M/16, grid.y = N/128.  Wt is [N][K] bf16 (pre-transposed).
// HAS_A2 templated so the K-loop has no per-iteration uniform branch.
template <bool HAS_A2>
__global__ void gemm_wmma_kernel(
    const float* __restrict__ A, const float* __restrict__ A2,
    const __bf16* __restrict__ Wt, const float* __restrict__ bias,
    const float* __restrict__ resid, float* __restrict__ outf,
    __bf16* __restrict__ outb,
    int M, int N, int K, float oscale, int relu, int vt_mode, int S, int Sp) {
  int lane = threadIdx.x & 31;
  int wv   = threadIdx.x >> 5;
  int m0 = blockIdx.x * 16;
  int n0 = blockIdx.y * 128 + wv * 32;
  int hi = lane >> 4;
  int c0 = hi ? 8 : 0, c1 = hi ? 24 : 16;
  int l15 = lane & 15;

  int arow = m0 + l15;
  const float* arp  = A + (size_t)arow * K;
  const float* arp2 = HAS_A2 ? A2 + (size_t)arow * K : nullptr;
  int ncol = n0 + l15;
  const __bf16* b0p = Wt + (size_t)ncol * K;
  const __bf16* b1p = Wt + (size_t)(ncol + 16) * K;

  v8f acc0 = {0.f, 0.f, 0.f, 0.f, 0.f, 0.f, 0.f, 0.f};
  v8f acc1 = {0.f, 0.f, 0.f, 0.f, 0.f, 0.f, 0.f, 0.f};

  // All K here are multiples of 64: unroll x2 so the scheduler overlaps two
  // load waves against two WMMA pairs; prefetch the next A K-tile.
#pragma unroll 2
  for (int kk = 0; kk < K; kk += 32) {
    __builtin_prefetch(arp + kk + 64, 0, 0);   // global_prefetch_b8
    v16bf af = HAS_A2 ? frag_from_f32_add(arp + kk, arp2 + kk, c0, c1)
                      : frag_from_f32(arp + kk, c0, c1);
    v16bf bf0 = frag_from_bf16(b0p + kk, c0, c1);
    v16bf bf1 = frag_from_bf16(b1p + kk, c0, c1);
    acc0 = WMMA_BF16(af, bf0, acc0);
    acc1 = WMMA_BF16(af, bf1, acc1);
  }

  float bias0 = bias[ncol];
  float bias1 = bias[ncol + 16];
#pragma unroll
  for (int j = 0; j < 8; ++j) {
    int row = m0 + j + hi * 8;
    float v0 = (acc0[j] + bias0) * oscale;
    float v1 = (acc1[j] + bias1) * oscale;
    if (relu) { v0 = fmaxf(v0, 0.f); v1 = fmaxf(v1, 0.f); }
    size_t i0 = (size_t)row * N + ncol;
    size_t i1 = i0 + 16;
    if (resid) { v0 += resid[i0]; v1 += resid[i1]; }
    if (outf)  { outf[i0] = v0;  outf[i1] = v1; }
    if (outb) {
      if (vt_mode) {  // [b][h][d][Sp] bf16 (attention V operand layout)
        int b = row / S, sidx = row - b * S;
        int h0 = ncol >> 5,        d0 = ncol & 31;
        int h1 = (ncol + 16) >> 5, d1 = (ncol + 16) & 31;
        outb[((size_t)(b * NHEADS + h0) * HEAD_DIM + d0) * Sp + sidx] = (__bf16)v0;
        outb[((size_t)(b * NHEADS + h1) * HEAD_DIM + d1) * Sp + sidx] = (__bf16)v1;
      } else {
        outb[i0] = (__bf16)v0;
        outb[i1] = (__bf16)v1;
      }
    }
  }
}

// ---- flash attention: one wave per (b, h, 16-query tile) ------------------
// qc [B*Q][256] bf16 (pre-scaled by ATTN_SCALE), kc [B*S][256] bf16,
// vt [b][h][d][Sp] bf16, mask [B][Q][S] f32 (templated), o [B*Q][256] f32.
// Main loop runs full 32-wide chunks with pure pointer-increment addressing;
// a single tail chunk handles S%32 with clamps/-inf masking.
template <bool HAS_MASK>
__global__ void attn_wmma_kernel(
    const __bf16* __restrict__ qc, const __bf16* __restrict__ kc,
    const __bf16* __restrict__ vt, const float* __restrict__ mask,
    float* __restrict__ o, int Bsz, int Q, int S, int Sp) {
  __shared__ __align__(16) __bf16 pbuf[8][16 * 32];  // per-wave P tile
  int lane = threadIdx.x & 31;
  int wv   = threadIdx.x >> 5;
  int nqt  = (Q + 15) >> 4;
  int unit = blockIdx.x * 8 + wv;
  int qt = unit % nqt;
  int h  = (unit / nqt) % NHEADS;
  int b  = unit / (nqt * NHEADS);
  int q0 = qt * 16;
  int hi = lane >> 4;
  int c0 = hi ? 8 : 0, c1 = hi ? 24 : 16;
  int l15 = lane & 15;

  int qrow = q0 + l15; if (qrow >= Q) qrow = Q - 1;
  v16bf qa = frag_from_bf16(qc + ((size_t)(b * Q + qrow)) * D_MODEL + h * HEAD_DIM, c0, c1);

  float mstat[8], lstat[8];
#pragma unroll
  for (int j = 0; j < 8; ++j) { mstat[j] = -1e30f; lstat[j] = 0.f; }
  v8f o0 = {0.f, 0.f, 0.f, 0.f, 0.f, 0.f, 0.f, 0.f};
  v8f o1 = {0.f, 0.f, 0.f, 0.f, 0.f, 0.f, 0.f, 0.f};
  __bf16* pb = &pbuf[wv][0];

  // strength-reduced operand pointers (advance by constants per chunk)
  const __bf16* kp0 = kc + ((size_t)b * S + l15) * D_MODEL + h * HEAD_DIM;
  const __bf16* kp1 = kp0 + (size_t)16 * D_MODEL;
  const __bf16* vbase = vt + (size_t)(b * NHEADS + h) * HEAD_DIM * Sp;
  const __bf16* vp0 = vbase + (size_t)l15 * Sp;
  const __bf16* vp1 = vbase + (size_t)(16 + l15) * Sp;
  const float* mp[8];
  if (HAS_MASK) {
#pragma unroll
    for (int j = 0; j < 8; ++j) {
      int rq = q0 + j + hi * 8; if (rq >= Q) rq = Q - 1;
      mp[j] = mask + ((size_t)b * Q + rq) * S + l15;
    }
  }

  // shared softmax-update + P@V body (x0/x1 already have mask/-inf applied)
  auto softmax_pv = [&](const float* x0, const float* x1,
                        const __bf16* vq0, const __bf16* vq1) {
#pragma unroll
    for (int j = 0; j < 8; ++j) {
      float rm = fmaxf(x0[j], x1[j]);
      rm = fmaxf(rm, __shfl_xor(rm, 1, 32));
      rm = fmaxf(rm, __shfl_xor(rm, 2, 32));
      rm = fmaxf(rm, __shfl_xor(rm, 4, 32));
      rm = fmaxf(rm, __shfl_xor(rm, 8, 32));
      float mnew  = fmaxf(mstat[j], rm);
      float alpha = __expf(mstat[j] - mnew);
      mstat[j] = mnew;
      float e0 = __expf(x0[j] - mnew);
      float e1 = __expf(x1[j] - mnew);
      float rs = e0 + e1;
      rs += __shfl_xor(rs, 1, 32);
      rs += __shfl_xor(rs, 2, 32);
      rs += __shfl_xor(rs, 4, 32);
      rs += __shfl_xor(rs, 8, 32);
      lstat[j] = lstat[j] * alpha + rs;
      o0[j] *= alpha;
      o1[j] *= alpha;
      int prow = j + hi * 8;
      pb[prow * 32 + l15]      = (__bf16)e0;   // C-layout -> LDS row-major
      pb[prow * 32 + 16 + l15] = (__bf16)e1;
    }
    asm volatile("s_wait_dscnt 0" ::: "memory");    // stores -> A-frag reads
    v16bf pa  = frag_from_bf16(pb + l15 * 32, c0, c1);  // ds_load_b128 pair
    v16bf vb0 = frag_from_bf16(vq0, c0, c1);
    v16bf vb1 = frag_from_bf16(vq1, c0, c1);
    o0 = WMMA_BF16(pa, vb0, o0);
    o1 = WMMA_BF16(pa, vb1, o1);
    asm volatile("s_wait_dscnt 0" ::: "memory");    // keep pbuf reuse ordered
  };

  int Smain = S & ~31;
  for (int s0 = 0; s0 < Smain; s0 += 32) {
    v16bf kb0 = frag_from_bf16(kp0, c0, c1);
    v16bf kb1 = frag_from_bf16(kp1, c0, c1);
    v8f z = {0.f, 0.f, 0.f, 0.f, 0.f, 0.f, 0.f, 0.f};
    v8f p0 = WMMA_BF16(qa, kb0, z);
    v8f p1 = WMMA_BF16(qa, kb1, z);
    float x0[8], x1[8];
#pragma unroll
    for (int j = 0; j < 8; ++j) {
      x0[j] = p0[j] + (HAS_MASK ? mp[j][0]  : 0.f);
      x1[j] = p1[j] + (HAS_MASK ? mp[j][16] : 0.f);
    }
    softmax_pv(x0, x1, vp0, vp1);
    kp0 += 32 * D_MODEL;
    kp1 += 32 * D_MODEL;
    vp0 += 32;
    vp1 += 32;
    if (HAS_MASK) {
#pragma unroll
      for (int j = 0; j < 8; ++j) mp[j] += 32;
    }
  }

  if (Smain < S) {  // single clamped tail chunk
    int s0 = Smain;
    int sA = s0 + l15;       if (sA >= S) sA = S - 1;
    int sB = s0 + 16 + l15;  if (sB >= S) sB = S - 1;
    v16bf kb0 = frag_from_bf16(kc + ((size_t)(b * S + sA)) * D_MODEL + h * HEAD_DIM, c0, c1);
    v16bf kb1 = frag_from_bf16(kc + ((size_t)(b * S + sB)) * D_MODEL + h * HEAD_DIM, c0, c1);
    v8f z = {0.f, 0.f, 0.f, 0.f, 0.f, 0.f, 0.f, 0.f};
    v8f p0 = WMMA_BF16(qa, kb0, z);
    v8f p1 = WMMA_BF16(qa, kb1, z);
    bool val0 = (s0 + l15) < S, val1 = (s0 + 16 + l15) < S;
    float x0[8], x1[8];
#pragma unroll
    for (int j = 0; j < 8; ++j) {
      float t0 = p0[j], t1 = p1[j];
      if (HAS_MASK) {
        t0 += val0 ? mp[j][0]  : 0.f;
        t1 += val1 ? mp[j][16] : 0.f;
      }
      x0[j] = val0 ? t0 : -1e30f;
      x1[j] = val1 ? t1 : -1e30f;
    }
    softmax_pv(x0, x1, vp0, vp1);   // vt is Sp-padded: reads stay in-bounds
  }

#pragma unroll
  for (int j = 0; j < 8; ++j) {
    int q = q0 + j + hi * 8;
    if (q < Q) {
      float inv = 1.0f / lstat[j];
      size_t base = ((size_t)(b * Q + q)) * D_MODEL + h * HEAD_DIM;
      o[base + l15]      = o0[j] * inv;
      o[base + 16 + l15] = o1[j] * inv;
    }
  }
}

// ---------------------------------------------------------------------------
extern "C" void kernel_launch(void* const* d_in, const int* in_sizes, int n_in,
                              void* d_out, int out_size, void* d_ws, size_t ws_size,
                              hipStream_t stream) {
  const int B = 16, Q = 100, S = 8192;
  const int M_q = B * Q;        // 1600
  const int M_kv = B * S;       // 131072
  const int Sp_self = 128;      // self-attn S=100 padded to 32

  // inputs (setup_inputs dict order; params nested in insertion order)
  const float* tgt   = (const float*)d_in[0];
  const float* mem   = (const float*)d_in[1];
  const float* qpos  = (const float*)d_in[2];
  const float* mpos  = (const float*)d_in[3];
  const float* cmask = (const float*)d_in[4];
  const float* cWq = (const float*)d_in[5];  const float* cbq = (const float*)d_in[6];
  const float* cWk = (const float*)d_in[7];  const float* cbk = (const float*)d_in[8];
  const float* cWv = (const float*)d_in[9];  const float* cbv = (const float*)d_in[10];
  const float* cWo = (const float*)d_in[11]; const float* cbo = (const float*)d_in[12];
  const float* sWq = (const float*)d_in[13]; const float* sbq = (const float*)d_in[14];
  const float* sWk = (const float*)d_in[15]; const float* sbk = (const float*)d_in[16];
  const float* sWv = (const float*)d_in[17]; const float* sbv = (const float*)d_in[18];
  const float* sWo = (const float*)d_in[19]; const float* sbo = (const float*)d_in[20];
  const float* g1 = (const float*)d_in[21];  const float* b1 = (const float*)d_in[22];
  const float* g2 = (const float*)d_in[23];  const float* b2 = (const float*)d_in[24];
  const float* g3 = (const float*)d_in[25];  const float* b3 = (const float*)d_in[26];
  const float* W1 = (const float*)d_in[27];  const float* bf1 = (const float*)d_in[28];
  const float* W2 = (const float*)d_in[29];  const float* bf2 = (const float*)d_in[30];

  // workspace carve-up (256B aligned)
  char* wsp = (char*)d_ws;
  auto carve = [&](size_t bytes) -> void* {
    void* p = (void*)wsp;
    wsp += (bytes + 255) & ~(size_t)255;
    return p;
  };
  __bf16* wt_cq = (__bf16*)carve(65536 * 2);
  __bf16* wt_ck = (__bf16*)carve(65536 * 2);
  __bf16* wt_cv = (__bf16*)carve(65536 * 2);
  __bf16* wt_co = (__bf16*)carve(65536 * 2);
  __bf16* wt_sq = (__bf16*)carve(65536 * 2);
  __bf16* wt_sk = (__bf16*)carve(65536 * 2);
  __bf16* wt_sv = (__bf16*)carve(65536 * 2);
  __bf16* wt_so = (__bf16*)carve(65536 * 2);
  __bf16* wt_f1 = (__bf16*)carve(262144 * 2);
  __bf16* wt_f2 = (__bf16*)carve(262144 * 2);
  float*  tn    = (float*)carve((size_t)M_q * 256 * 4);
  __bf16* qcb   = (__bf16*)carve((size_t)M_q * 256 * 2);
  __bf16* kcs   = (__bf16*)carve((size_t)M_q * 256 * 2);
  float*  obuf  = (float*)carve((size_t)M_q * 256 * 4);
  float*  tgt1  = (float*)carve((size_t)M_q * 256 * 4);
  float*  tgt2  = (float*)carve((size_t)M_q * 256 * 4);
  float*  hbuf  = (float*)carve((size_t)M_q * 1024 * 4);
  __bf16* vts   = (__bf16*)carve((size_t)B * NHEADS * HEAD_DIM * Sp_self * 2);
  __bf16* kcb   = (__bf16*)carve((size_t)M_kv * 256 * 2);
  __bf16* vtb   = (__bf16*)carve((size_t)M_kv * 256 * 2);
  (void)ws_size; (void)n_in; (void)in_sizes; (void)out_size;

  // 1) weight transposes (f32 [K][N] -> bf16 [N][K])
  auto wt = [&](const float* W, __bf16* Wt, int K, int N) {
    int tot = K * N;
    wtrans_kernel<<<(tot + 255) / 256, 256, 0, stream>>>(W, Wt, K, N);
  };
  wt(cWq, wt_cq, 256, 256); wt(cWk, wt_ck, 256, 256);
  wt(cWv, wt_cv, 256, 256); wt(cWo, wt_co, 256, 256);
  wt(sWq, wt_sq, 256, 256); wt(sWk, wt_sk, 256, 256);
  wt(sWv, wt_sv, 256, 256); wt(sWo, wt_so, 256, 256);
  wt(W1, wt_f1, 256, 1024); wt(W2, wt_f2, 1024, 256);

  dim3 blk(128);
  dim3 gQ(M_q / 16, 2);       // 1600x256 gemms
  dim3 gKV(M_kv / 16, 2);     // 131072x256 gemms
  dim3 gF1(M_q / 16, 8);      // 1600x1024

  // ---- cross attention -----------------------------------------------------
  ln_kernel<<<M_q, 256, 0, stream>>>(tgt, g1, b1, tn);
  // q = ((LN(tgt)+qpos) @ Wq + bq) * scale -> bf16
  gemm_wmma_kernel<true><<<gQ, blk, 0, stream>>>(tn, qpos, wt_cq, cbq, nullptr, nullptr,
                                                 qcb, M_q, 256, 256, ATTN_SCALE, 0, 0, 1, 1);
  // k = (mem+mpos) @ Wk + bk -> bf16 [B*S][256]
  gemm_wmma_kernel<true><<<gKV, blk, 0, stream>>>(mem, mpos, wt_ck, cbk, nullptr, nullptr,
                                                  kcb, M_kv, 256, 256, 1.0f, 0, 0, 1, 1);
  // v = mem @ Wv + bv -> bf16 transposed [b][h][d][S]
  gemm_wmma_kernel<false><<<gKV, blk, 0, stream>>>(mem, nullptr, wt_cv, cbv, nullptr,
                                                   nullptr, vtb, M_kv, 256, 256, 1.0f,
                                                   0, 1, S, S);
  // attention: 16*8*7 = 896 wave-units = 112 blocks x 8 waves
  attn_wmma_kernel<true><<<112, 256, 0, stream>>>(qcb, kcb, vtb, cmask, obuf, B, Q, S, S);
  // tgt1 = tgt + attn_out @ Wo + bo
  gemm_wmma_kernel<false><<<gQ, blk, 0, stream>>>(obuf, nullptr, wt_co, cbo, tgt, tgt1,
                                                  nullptr, M_q, 256, 256, 1.0f, 0, 0, 1, 1);

  // ---- self attention ------------------------------------------------------
  ln_kernel<<<M_q, 256, 0, stream>>>(tgt1, g2, b2, tn);
  gemm_wmma_kernel<true><<<gQ, blk, 0, stream>>>(tn, qpos, wt_sq, sbq, nullptr, nullptr,
                                                 qcb, M_q, 256, 256, ATTN_SCALE, 0, 0, 1, 1);
  gemm_wmma_kernel<true><<<gQ, blk, 0, stream>>>(tn, qpos, wt_sk, sbk, nullptr, nullptr,
                                                 kcs, M_q, 256, 256, 1.0f, 0, 0, 1, 1);
  gemm_wmma_kernel<false><<<gQ, blk, 0, stream>>>(tn, nullptr, wt_sv, sbv, nullptr, nullptr,
                                                  vts, M_q, 256, 256, 1.0f, 0, 1, Q, Sp_self);
  attn_wmma_kernel<false><<<112, 256, 0, stream>>>(qcb, kcs, vts, nullptr, obuf,
                                                   B, Q, Q, Sp_self);
  gemm_wmma_kernel<false><<<gQ, blk, 0, stream>>>(obuf, nullptr, wt_so, sbo, tgt1, tgt2,
                                                  nullptr, M_q, 256, 256, 1.0f, 0, 0, 1, 1);

  // ---- FFN -----------------------------------------------------------------
  ln_kernel<<<M_q, 256, 0, stream>>>(tgt2, g3, b3, tn);
  gemm_wmma_kernel<false><<<gF1, blk, 0, stream>>>(tn, nullptr, wt_f1, bf1, nullptr, hbuf,
                                                   nullptr, M_q, 1024, 256, 1.0f, 1, 0, 1, 1);
  gemm_wmma_kernel<false><<<gQ, blk, 0, stream>>>(hbuf, nullptr, wt_f2, bf2, tgt2,
                                                  (float*)d_out, nullptr,
                                                  M_q, 256, 1024, 1.0f, 0, 0, 1, 1);
}
